// MultiHeadAttention_16569983828599
// MI455X (gfx1250) — matrix operand order repaired and checked by reading, not compiled, not run on hardware
//
#include <hip/hip_runtime.h>
#include <hip/hip_bf16.h>
#include <stdint.h>

// ---------------------------------------------------------------------------
// MI455X (gfx1250) fused multi-head attention, f16 WMMA path.
//   ~69 GFLOP total vs ~150MB traffic @ 23.3 TB/s -> compute bound -> all four
//   matmul stages run on v_wmma_f32_16x16x32_f16 (f32 accumulate).
//   f16 staging copies use GLOBAL_LOAD_ASYNC_TO_LDS_B128 (ASYNCcnt);
//   global_prefetch_b8 warms the next tile a full compute phase ahead.
// ---------------------------------------------------------------------------

typedef __attribute__((ext_vector_type(16))) _Float16 v16h;
typedef __attribute__((ext_vector_type(8)))  float    v8f;
typedef __attribute__((ext_vector_type(4)))  int      v4i;

#if defined(__has_builtin)
#if __has_builtin(__builtin_amdgcn_global_load_async_to_lds_b128)
#define ASYNC_COPY 1
#endif
#endif

#ifdef ASYNC_COPY
typedef __attribute__((address_space(1))) v4i gv4i;   // global int4
typedef __attribute__((address_space(3))) v4i lv4i;   // LDS int4
#define ASYNC_B128(gp, lp) \
  __builtin_amdgcn_global_load_async_to_lds_b128((gv4i*)(gp), (lv4i*)(lp), 0, 0)
#define ASYNC_WAIT() asm volatile("s_wait_asynccnt 0x0" ::: "memory")
#else
#define ASYNC_WAIT()
#endif

union FragH { v16h h; uint32_t u[8]; };

// A fragment, 16x32 f16, from row-major [M][K] (K contiguous), per ISA 7.12.2:
// lanes 0-15: M=lane, K 0..7 in v0..3 (pairs), 16..23 in v4..7
// lanes 16-31: M=lane-16, K 8..15 / 24..31
__device__ __forceinline__ v16h load_a_frag(const _Float16* base, int strideM, int lane) {
  int row = lane & 15;
  int hi  = lane >> 4;
  const _Float16* p = base + row * strideM + hi * 8;
  FragH f;
#pragma unroll
  for (int v = 0; v < 8; ++v) {
    int kb = ((v & 3) * 2) + ((v >> 2) * 16);
    f.u[v] = *(const uint32_t*)(p + kb);
  }
  return f.h;
}

// B fragment, 32x16 f16, from [N][K] storage (K contiguous per row of N):
// lanes 0-15: N=lane, K 0..7 / 16..23; lanes 16-31: N=lane-16, K 8..15 / 24..31
__device__ __forceinline__ v16h load_b_frag(const _Float16* base, int strideN, int lane) {
  int col = lane & 15;
  int kg  = lane >> 4;
  const _Float16* p = base + col * strideN + kg * 8;
  FragH f;
#pragma unroll
  for (int v = 0; v < 8; ++v) f.u[v] = *(const uint32_t*)(p + ((v & 3) * 2) + ((v >> 2) * 16));
  return f.h;
}

__device__ __forceinline__ v8f wmma_f16(v16h a, v16h b, v8f c) {
  return __builtin_amdgcn_wmma_f32_16x16x32_f16(false, a, false, b, (short)0, c,
                                                false, false);
}

// ---------------------------------------------------------------------------
// Kernel 1: fused QKV projection.  X[4096,1024](f32) @ W[1024,1024](f32) + b
// -> f16 output permuted to [B][H][S][Hd] head-major (so attention's K tile is
// already in B-fragment layout).  grid = (32, 8, 3); block = 256 (8 waves).
// Block tile 128x128, wave tile 32x64 (2x4 of 16x16), K step 32.
// (f32->f16 conversion happens in the stage, so this one keeps VALU staging.)
// ---------------------------------------------------------------------------
__global__ __launch_bounds__(256) void qkv_proj_kernel(
    const float* __restrict__ qin, const float* __restrict__ kin,
    const float* __restrict__ vin,
    const float* __restrict__ Wq, const float* __restrict__ Wk,
    const float* __restrict__ Wv,
    const float* __restrict__ bq, const float* __restrict__ bk,
    const float* __restrict__ bv,
    _Float16* __restrict__ out) {
  __shared__ _Float16 ldsA[128 * 32];   // [M][K], K contiguous
  __shared__ _Float16 ldsB[128 * 34];   // [N][K] transposed, padded stride

  const int z = blockIdx.z;
  const float* X    = (z == 0) ? qin : (z == 1) ? kin : vin;
  const float* W    = (z == 0) ? Wq  : (z == 1) ? Wk  : Wv;
  const float* bias = (z == 0) ? bq  : (z == 1) ? bk  : bv;
  _Float16* O = out + (size_t)z * (4096u * 1024u);

  const int tid = threadIdx.x, lane = tid & 31, wid = tid >> 5;
  const int wm = wid >> 1, wn = wid & 1;
  const int m0 = blockIdx.x * 128, n0 = blockIdx.y * 128;

  v8f acc[2][4] = {};

  for (int kt = 0; kt < 32; ++kt) {
    const int k0 = kt * 32;
    // stage A: 128x32 f32 -> f16 (coalesced float4 reads)
#pragma unroll
    for (int i = 0; i < 4; ++i) {
      int f = i * 1024 + tid * 4;
      int row = f >> 5, kk = f & 31;
      float4 x = *(const float4*)(X + (size_t)(m0 + row) * 1024 + k0 + kk);
      _Float16* d = ldsA + row * 32 + kk;
      d[0] = (_Float16)x.x; d[1] = (_Float16)x.y;
      d[2] = (_Float16)x.z; d[3] = (_Float16)x.w;
    }
    // stage B transposed: W row-major [K][N] -> lds [N][K]
#pragma unroll
    for (int i = 0; i < 4; ++i) {
      int f = i * 1024 + tid * 4;
      int kk = f >> 7, nn = f & 127;
      float4 w = *(const float4*)(W + (size_t)(k0 + kk) * 1024 + n0 + nn);
      ldsB[(nn + 0) * 34 + kk] = (_Float16)w.x;
      ldsB[(nn + 1) * 34 + kk] = (_Float16)w.y;
      ldsB[(nn + 2) * 34 + kk] = (_Float16)w.z;
      ldsB[(nn + 3) * 34 + kk] = (_Float16)w.w;
    }
    // prefetch next K-tile (global_prefetch_b8; whole cacheline per lane)
    if (kt + 1 < 32) {
      int nk = k0 + 32;
      if (tid < 128) __builtin_prefetch(X + (size_t)(m0 + tid) * 1024 + nk);
      if (tid < 32)  __builtin_prefetch(W + (size_t)(nk + tid) * 1024 + n0);
      if (tid >= 32 && tid < 64)
        __builtin_prefetch(W + (size_t)(nk + tid - 32) * 1024 + n0 + 64);
    }
    __syncthreads();

    v16h a0 = load_a_frag(ldsA + (wm * 32 +  0) * 32, 32, lane);
    v16h a1 = load_a_frag(ldsA + (wm * 32 + 16) * 32, 32, lane);
#pragma unroll
    for (int j = 0; j < 4; ++j) {
      v16h b = load_b_frag(ldsB + (wn * 64 + j * 16) * 34, 34, lane);
      acc[0][j] = wmma_f16(a0, b, acc[0][j]);
      acc[1][j] = wmma_f16(a1, b, acc[1][j]);
    }
    __syncthreads();
  }

  // epilogue: + bias, f16 store to [B][H][S][Hd]
  const int col = lane & 15, hi = lane >> 4;
#pragma unroll
  for (int i = 0; i < 2; ++i) {
#pragma unroll
    for (int j = 0; j < 4; ++j) {
      int n = n0 + wn * 64 + j * 16 + col;
      float bb = bias[n];
      int h = n >> 6, hd = n & 63;
#pragma unroll
      for (int r = 0; r < 8; ++r) {
        int m = m0 + wm * 32 + i * 16 + hi * 8 + r;
        int b = m >> 11, s = m & 2047;
        O[(((size_t)(b * 16 + h)) * 2048 + s) * 64 + hd] =
            (_Float16)(acc[i][j][r] + bb);
      }
    }
  }
}

// ---------------------------------------------------------------------------
// Kernel 2: flash attention (no mask).  grid = (S/64, B*H); block = 128.
// Each wave owns 16 query rows; Q fragments (pre-scaled by 1/sqrt(64)=0.125)
// stay in registers across the whole key loop.  Key blocks of 64 staged in
// LDS (K via async-to-LDS); online softmax; P staged through per-wave LDS to
// convert C-layout -> A-layout for the P@V WMMAs.
// ---------------------------------------------------------------------------
__global__ __launch_bounds__(128) void attention_kernel(
    const _Float16* __restrict__ Qw, const _Float16* __restrict__ Kw,
    const _Float16* __restrict__ Vw, _Float16* __restrict__ attn) {
  __shared__ _Float16 ldsK [64 * 72];       // [key][hd]  (== B layout [N][K])
  __shared__ _Float16 ldsVt[64 * 72];       // [hd][key]  (B layout for P@V)
  __shared__ _Float16 ldsP [4][16 * 64];    // per-wave P scratch [q][key]

  const int tid = threadIdx.x, lane = tid & 31, wid = tid >> 5;
  const int bh = blockIdx.y;                 // b*16 + h
  const int q0 = blockIdx.x * 64;

  // Q fragments: 16 rows x 64 hd, two K-steps, loaded once from global.
  const _Float16* Qb = Qw + ((size_t)bh * 2048 + q0 + wid * 16) * 64;
  v16h aQ0 = load_a_frag(Qb,      64, lane);
  v16h aQ1 = load_a_frag(Qb + 32, 64, lane);
#pragma unroll
  for (int e = 0; e < 16; ++e) {            // fold 1/SCALE = 0.125 (exact)
    aQ0[e] = aQ0[e] * (_Float16)0.125f;
    aQ1[e] = aQ1[e] * (_Float16)0.125f;
  }

  float m[8], l[8];
#pragma unroll
  for (int r = 0; r < 8; ++r) { m[r] = -3.0e38f; l[r] = 0.0f; }
  v8f O[4] = {};

  const int col = lane & 15, hi = lane >> 4;

  for (int kb = 0; kb < 32; ++kb) {
    const _Float16* Kg = Kw + ((size_t)bh * 2048 + kb * 64) * 64;
    const _Float16* Vg = Vw + ((size_t)bh * 2048 + kb * 64) * 64;
    // stage K block: pure f16 copy -> async DMA to LDS
#pragma unroll
    for (int i = 0; i < 4; ++i) {
      int idx = i * 128 + tid;
      int row = idx >> 3, ko = (idx & 7) * 8;
#ifdef ASYNC_COPY
      ASYNC_B128(Kg + row * 64 + ko, ldsK + row * 72 + ko);
#else
      *(uint4*)(ldsK + row * 72 + ko) = *(const uint4*)(Kg + row * 64 + ko);
#endif
    }
    // stage V block transposed: [s][hd] -> [hd][s] (needs VGPR round-trip)
#pragma unroll
    for (int i = 0; i < 8; ++i) {
      int idx = i * 128 + tid;
      int s = idx >> 4, hd = (idx & 15) * 4;
      uint2 d = *(const uint2*)(Vg + s * 64 + hd);
      _Float16 t[4]; *(uint2*)t = d;
      ldsVt[(hd + 0) * 72 + s] = t[0];
      ldsVt[(hd + 1) * 72 + s] = t[1];
      ldsVt[(hd + 2) * 72 + s] = t[2];
      ldsVt[(hd + 3) * 72 + s] = t[3];
    }
    // prefetch next key block while this one is being consumed
    if (kb + 1 < 32 && tid < 64) {
      __builtin_prefetch(Kg + 64 * 64 + tid * 64);
      __builtin_prefetch(Vg + 64 * 64 + tid * 64);
    }
    ASYNC_WAIT();
    __syncthreads();

    // S = (Q/sqrt(d)) @ K^T : 4 key-subtiles x 2 K-steps = 8 WMMAs
    v8f S[4];
#pragma unroll
    for (int nt = 0; nt < 4; ++nt) {
      v8f s = {};
      s = wmma_f16(aQ0, load_b_frag(ldsK + nt * 16 * 72,      72, lane), s);
      s = wmma_f16(aQ1, load_b_frag(ldsK + nt * 16 * 72 + 32, 72, lane), s);
      S[nt] = s;
    }

    // online softmax: row reductions over the 16-lane N groups
    float alpha[8];
#pragma unroll
    for (int r = 0; r < 8; ++r) {
      float bm = fmaxf(fmaxf(S[0][r], S[1][r]), fmaxf(S[2][r], S[3][r]));
      bm = fmaxf(bm, __shfl_xor(bm, 1));
      bm = fmaxf(bm, __shfl_xor(bm, 2));
      bm = fmaxf(bm, __shfl_xor(bm, 4));
      bm = fmaxf(bm, __shfl_xor(bm, 8));
      float mn = fmaxf(m[r], bm);
      alpha[r] = __expf(m[r] - mn);
      m[r] = mn;
    }
    float rs[8];
#pragma unroll
    for (int r = 0; r < 8; ++r) rs[r] = 0.0f;
#pragma unroll
    for (int nt = 0; nt < 4; ++nt) {
#pragma unroll
      for (int r = 0; r < 8; ++r) {
        float p = __expf(S[nt][r] - m[r]);
        S[nt][r] = p;
        rs[r] += p;
      }
    }
#pragma unroll
    for (int r = 0; r < 8; ++r) {
      rs[r] += __shfl_xor(rs[r], 1);
      rs[r] += __shfl_xor(rs[r], 2);
      rs[r] += __shfl_xor(rs[r], 4);
      rs[r] += __shfl_xor(rs[r], 8);
      l[r] = l[r] * alpha[r] + rs[r];
    }

    // C-layout -> A-layout transpose of P via per-wave LDS
    _Float16* P = ldsP[wid];
#pragma unroll
    for (int nt = 0; nt < 4; ++nt)
#pragma unroll
      for (int r = 0; r < 8; ++r)
        P[(hi * 8 + r) * 64 + nt * 16 + col] = (_Float16)S[nt][r];
    asm volatile("s_wait_dscnt 0x0" ::: "memory");  // cross-lane LDS RAW

    // O = alpha*O + P @ V : 2 K-steps x 4 hd-subtiles = 8 WMMAs
#pragma unroll
    for (int nt = 0; nt < 4; ++nt)
#pragma unroll
      for (int r = 0; r < 8; ++r) O[nt][r] *= alpha[r];
#pragma unroll
    for (int ks = 0; ks < 2; ++ks) {
      v16h aP = load_a_frag(P + ks * 32, 64, lane);
#pragma unroll
      for (int nt = 0; nt < 4; ++nt)
        O[nt] = wmma_f16(
            aP, load_b_frag(ldsVt + nt * 16 * 72 + ks * 32, 72, lane), O[nt]);
    }
    __syncthreads();
  }

  // finalize: O / l, f16 store to [B*S][D] for the output projection
  const int b = bh >> 4, h = bh & 15;
#pragma unroll
  for (int nt = 0; nt < 4; ++nt) {
#pragma unroll
    for (int r = 0; r < 8; ++r) {
      int s = q0 + wid * 16 + hi * 8 + r;
      float o = O[nt][r] / l[r];
      attn[((size_t)(b * 2048 + s)) * 1024 + h * 64 + nt * 16 + col] =
          (_Float16)o;
    }
  }
}

// ---------------------------------------------------------------------------
// Kernel 3: output projection.  attn[4096,1024](f16) @ Wo + bo -> f32 d_out.
// Same tiling as kernel 1; A tile is already f16 -> async-to-LDS copy.
// ---------------------------------------------------------------------------
__global__ __launch_bounds__(256) void out_proj_kernel(
    const _Float16* __restrict__ X, const float* __restrict__ W,
    const float* __restrict__ bias, float* __restrict__ out) {
  __shared__ _Float16 ldsA[128 * 32];
  __shared__ _Float16 ldsB[128 * 34];

  const int tid = threadIdx.x, lane = tid & 31, wid = tid >> 5;
  const int wm = wid >> 1, wn = wid & 1;
  const int m0 = blockIdx.x * 128, n0 = blockIdx.y * 128;

  v8f acc[2][4] = {};

  for (int kt = 0; kt < 32; ++kt) {
    const int k0 = kt * 32;
#pragma unroll
    for (int i = 0; i < 2; ++i) {
      int idx = i * 256 + tid;
      int row = idx >> 2, ko = (idx & 3) * 8;
#ifdef ASYNC_COPY
      ASYNC_B128(X + (size_t)(m0 + row) * 1024 + k0 + ko,
                 ldsA + row * 32 + ko);
#else
      *(uint4*)(ldsA + row * 32 + ko) =
          *(const uint4*)(X + (size_t)(m0 + row) * 1024 + k0 + ko);
#endif
    }
#pragma unroll
    for (int i = 0; i < 4; ++i) {
      int f = i * 1024 + tid * 4;
      int kk = f >> 7, nn = f & 127;
      float4 w = *(const float4*)(W + (size_t)(k0 + kk) * 1024 + n0 + nn);
      ldsB[(nn + 0) * 34 + kk] = (_Float16)w.x;
      ldsB[(nn + 1) * 34 + kk] = (_Float16)w.y;
      ldsB[(nn + 2) * 34 + kk] = (_Float16)w.z;
      ldsB[(nn + 3) * 34 + kk] = (_Float16)w.w;
    }
    if (kt + 1 < 32) {
      int nk = k0 + 32;
      if (tid < 128) __builtin_prefetch(X + (size_t)(m0 + tid) * 1024 + nk);
      if (tid < 32)  __builtin_prefetch(W + (size_t)(nk + tid) * 1024 + n0);
      if (tid >= 32 && tid < 64)
        __builtin_prefetch(W + (size_t)(nk + tid - 32) * 1024 + n0 + 64);
    }
    ASYNC_WAIT();
    __syncthreads();

    v16h a0 = load_a_frag(ldsA + (wm * 32 +  0) * 32, 32, lane);
    v16h a1 = load_a_frag(ldsA + (wm * 32 + 16) * 32, 32, lane);
#pragma unroll
    for (int j = 0; j < 4; ++j) {
      v16h b = load_b_frag(ldsB + (wn * 64 + j * 16) * 34, 34, lane);
      acc[0][j] = wmma_f16(a0, b, acc[0][j]);
      acc[1][j] = wmma_f16(a1, b, acc[1][j]);
    }
    __syncthreads();
  }

  const int col = lane & 15, hi = lane >> 4;
#pragma unroll
  for (int i = 0; i < 2; ++i) {
#pragma unroll
    for (int j = 0; j < 4; ++j) {
      int n = n0 + wn * 64 + j * 16 + col;
      float bb = bias[n];
#pragma unroll
      for (int r = 0; r < 8; ++r) {
        int mm = m0 + wm * 32 + i * 16 + hi * 8 + r;
        out[(size_t)mm * 1024 + n] = acc[i][j][r] + bb;
      }
    }
  }
}

// ---------------------------------------------------------------------------
extern "C" void kernel_launch(void* const* d_in, const int* in_sizes, int n_in,
                              void* d_out, int out_size, void* d_ws,
                              size_t ws_size, hipStream_t stream) {
  (void)in_sizes; (void)n_in; (void)out_size; (void)ws_size;
  const float* q  = (const float*)d_in[0];
  const float* k  = (const float*)d_in[1];
  const float* v  = (const float*)d_in[2];
  const float* Wq = (const float*)d_in[3];
  const float* bq = (const float*)d_in[4];
  const float* Wk = (const float*)d_in[5];
  const float* bk = (const float*)d_in[6];
  const float* Wv = (const float*)d_in[7];
  const float* bv = (const float*)d_in[8];
  const float* Wo = (const float*)d_in[9];
  const float* bo = (const float*)d_in[10];

  const size_t PER = 4096u * 1024u;          // elems per f16 tensor
  _Float16* qkv  = (_Float16*)d_ws;          // Q | K | V (head-major)
  _Float16* attn = qkv + 3 * PER;            // attention output, f16

  dim3 g1(32, 8, 3);                         // M/128, N/128, {Q,K,V}
  qkv_proj_kernel<<<g1, 256, 0, stream>>>(q, k, v, Wq, Wk, Wv, bq, bk, bv, qkv);

  dim3 g2(32, 32, 1);                        // S/64 query blocks, B*H
  attention_kernel<<<g2, 128, 0, stream>>>(qkv, qkv + PER, qkv + 2 * PER, attn);

  dim3 g3(32, 8, 1);
  out_proj_kernel<<<g3, 256, 0, stream>>>(attn, Wo, bo, (float*)d_out);
}
